// downprompt_34359738984
// MI455X (gfx1250) — compile-verified, matrix operand order checked by reading, not run on hardware
//
#include <hip/hip_runtime.h>
#include <hip/hip_bf16.h>
#include <math.h>

// ---------------- types for CDNA5 WMMA ----------------
typedef __attribute__((ext_vector_type(16))) __bf16 v16bf;
typedef __attribute__((ext_vector_type(8)))  __bf16 bf8;
typedef __attribute__((ext_vector_type(8)))  float  v8f;

// ---------------- problem sizes ----------------
constexpr int    kN    = 8192;
constexpr int    kD    = 50;
constexpr int    kDp   = 64;     // padded D
constexpr int    kFT   = 256;
constexpr int    kNIDX = 4096;
constexpr int    kC    = 10;

// ---------------- workspace layout (bytes) ----------------
constexpr size_t OFF_COEF = 0;                                   // 2*64 f32
constexpr size_t OFF_F1   = 512;                                 // N*Dp f32   = 2 MB
constexpr size_t OFF_BT   = OFF_F1   + (size_t)kN * kDp * 4;     // bf16 [128][N] = 2 MB
constexpr size_t OFF_HT   = OFF_BT   + (size_t)128 * kN * 2;     // bf16 [256][N] = 4 MB
constexpr size_t OFF_Y    = OFF_HT   + (size_t)kFT * kN * 2;     // f32 [N][128]  = 4 MB
constexpr size_t OFF_E    = OFF_Y    + (size_t)kN * 128 * 4;     // f32 [N][256]  = 8 MB
constexpr size_t OFF_SUMS = OFF_E    + (size_t)kN * kFT * 4;     // f32 [10][256]
constexpr size_t OFF_CNT  = OFF_SUMS + (size_t)kC * kFT * 4;     // f32 [16]
constexpr size_t OFF_PN   = OFF_CNT  + 64;                       // f32 [10][256]
constexpr size_t OFF_ACC  = OFF_PN   + (size_t)kC * kFT * 4;     // f32 [8]

__device__ __forceinline__ float elu_f(float x) {
    return x > 0.f ? x : __expf(x) - 1.f;
}

// ================= K0: prompt MLP -> per-column coefficients =================
// features1[i,d] = elu(coef_a[d] + coef_b[d] * features[i,d])
__global__ void prompt_kernel(const float* __restrict__ texttoken,
                              const float* __restrict__ wp,
                              const float* __restrict__ r_w1,
                              const float* __restrict__ r_b1,
                              const float* __restrict__ r_w2,
                              const float* __restrict__ r_b2,
                              const float* __restrict__ down,
                              const float* __restrict__ comb,
                              float* __restrict__ coef) {
    __shared__ float ge[4][64];
    __shared__ float red[64];
    __shared__ float guide[4];
    const int t = threadIdx.x; // 0..63
    #pragma unroll
    for (int tt = 0; tt < 4; ++tt)
        ge[tt][t] = (t < kD) ? wp[tt] * texttoken[tt * kD + t] : 0.f;
    __syncthreads();
    float part[4];
    #pragma unroll
    for (int tt = 0; tt < 4; ++tt) {
        float s = r_b1[t];
        for (int d = 0; d < kD; ++d) s += ge[tt][d] * r_w1[d * 64 + t];
        s = fmaxf(s, 0.f);
        part[tt] = s * r_w2[t];
    }
    for (int tt = 0; tt < 4; ++tt) {
        red[t] = part[tt];
        __syncthreads();
        for (int off = 32; off > 0; off >>= 1) {
            if (t < off) red[t] += red[t + off];
            __syncthreads();
        }
        if (t == 0) guide[tt] = red[0] + r_b2[0];
        __syncthreads();
    }
    if (t == 0) {
        float mx = guide[0];
        for (int tt = 1; tt < 4; ++tt) mx = fmaxf(mx, guide[tt]);
        float sm = 0.f;
        for (int tt = 0; tt < 4; ++tt) { guide[tt] = __expf(guide[tt] - mx); sm += guide[tt]; }
        for (int tt = 0; tt < 4; ++tt) guide[tt] /= sm;
    }
    __syncthreads();
    float tok = 0.f;
    #pragma unroll
    for (int tt = 0; tt < 4; ++tt) tok += guide[tt] * ge[tt][t];
    const float c0 = comb[0], c1 = comb[1];
    coef[t]      = (t < kD) ? c0 * tok : 0.f;
    coef[64 + t] = (t < kD) ? (c0 + c1 * down[t]) : 0.f;
}

// ======== K1: features1 (row-major, padded) + transposed bf16 B for GEMM1 ========
// Bt rows 0..63  : features1 column d (bf16),  Bt rows 64..127 : raw features column d
__global__ __launch_bounds__(256)
void features1_kernel(const float* __restrict__ features,
                      const float* __restrict__ coef,
                      float* __restrict__ F1,
                      __bf16* __restrict__ Bt) {
    __shared__ __bf16 t1[64][72];
    __shared__ __bf16 t0[64][72];
    const int i0 = blockIdx.x * 64;
    const int t = threadIdx.x;
    #pragma unroll
    for (int rep = 0; rep < 16; ++rep) {
        int id = rep * 256 + t;
        int il = id >> 6, d = id & 63;
        float f = (d < kD) ? features[(size_t)(i0 + il) * kD + d] : 0.f;
        float z = coef[d] + coef[64 + d] * f;
        float f1 = (d < kD) ? elu_f(z) : 0.f;
        F1[(size_t)(i0 + il) * kDp + d] = f1;
        t1[il][d] = (__bf16)f1;
        t0[il][d] = (__bf16)((d < kD) ? f : 0.f);
    }
    __syncthreads();
    #pragma unroll
    for (int rep = 0; rep < 2; ++rep) {
        int vec = rep * 256 + t;       // 0..511
        int d = vec >> 3;
        int i8 = (vec & 7) * 8;
        __bf16* o1 = Bt + (size_t)d * kN + i0 + i8;
        __bf16* o0 = Bt + (size_t)(64 + d) * kN + i0 + i8;
        #pragma unroll
        for (int x = 0; x < 8; ++x) { o1[x] = t1[i8 + x][d]; o0[x] = t0[i8 + x][d]; }
    }
}

// ========== K2: Hgcn^T (bf16) = (features1 @ w_gcn)^T, pre-transposed B for GEMM2 ==========
__global__ __launch_bounds__(256)
void hgcn_kernel(const float* __restrict__ F1,
                 const float* __restrict__ w_gcn,
                 __bf16* __restrict__ Ht) {
    __shared__ float sf[64][65];
    const int i0 = blockIdx.x * 64;
    const int f0 = blockIdx.y * 64;
    const int t = threadIdx.x;
    #pragma unroll
    for (int rep = 0; rep < 16; ++rep) {
        int id = rep * 256 + t;
        int il = id >> 6, d = id & 63;
        sf[il][d] = F1[(size_t)(i0 + il) * kDp + d];
    }
    __syncthreads();
    const int fl = t >> 2;
    const int iq = (t & 3) * 16;
    float acc[16];
    #pragma unroll
    for (int x = 0; x < 16; ++x) acc[x] = 0.f;
    for (int d = 0; d < kD; ++d) {
        float wv = w_gcn[d * kFT + f0 + fl];
        #pragma unroll
        for (int x = 0; x < 16; ++x) acc[x] += sf[iq + x][d] * wv;
    }
    __bf16* out = Ht + (size_t)(f0 + fl) * kN + i0 + iq;
    #pragma unroll
    for (int x = 0; x < 16; ++x) out[x] = (__bf16)acc[x];
}

// ================= K3: bf16 WMMA GEMM, A fp32 streamed, B^T bf16 in L2 =================
// C[m][n] = sum_k A[m][k] * Bt[n][k];  mode 0: *= eival[m];  mode 1: elu
__global__ __launch_bounds__(256)
void gemm_bf16_kernel(const float* __restrict__ A,
                      const __bf16* __restrict__ Bt,
                      float* __restrict__ Cout,
                      int ncols,
                      const float* __restrict__ eival,
                      int mode) {
    constexpr int AS_STRIDE = 40;                    // bf16 elems; 80B rows, bank-conflict-free
    __shared__ __bf16 As[128 * AS_STRIDE];
    const int tid  = threadIdx.x;
    const int lane = tid & 31;
    const int w    = tid >> 5;
    const int wm   = w & 3;            // 4 waves along M (32 rows each)
    const int wn   = w >> 2;           // 2 waves along N (32 cols each)
    const int mblk = blockIdx.x * 128;
    const int nblk = blockIdx.y * 64;

    const int arow  = tid >> 1;        // 0..127
    const int acol  = (tid & 1) * 16;  // 0 or 16
    const int l15   = lane & 15;
    const int khalf = (lane >> 4) * 8; // 0 or 8  (ISA 16-bit A/B k-mapping)

    v8f acc[2][2];
    #pragma unroll
    for (int a = 0; a < 2; ++a)
        #pragma unroll
        for (int b = 0; b < 2; ++b)
            #pragma unroll
            for (int r = 0; r < 8; ++r) acc[a][b][r] = 0.f;

    for (int k0 = 0; k0 < kN; k0 += 32) {
        // cooperative, coalesced A tile load (128 x 32 f32) -> bf16 in LDS
        const float* ap = A + (size_t)(mblk + arow) * kN + k0 + acol;
        __builtin_prefetch(ap + 64, 0, 1);           // global_prefetch for the streamed matrix
        __bf16* dst = &As[arow * AS_STRIDE + acol];
        #pragma unroll
        for (int q = 0; q < 4; ++q) {
            float4 v = *(const float4*)(ap + 4 * q);
            dst[4 * q + 0] = (__bf16)v.x;
            dst[4 * q + 1] = (__bf16)v.y;
            dst[4 * q + 2] = (__bf16)v.z;
            dst[4 * q + 3] = (__bf16)v.w;
        }
        __syncthreads();

        v16bf afrag[2], bfrag[2];
        #pragma unroll
        for (int mt = 0; mt < 2; ++mt) {
            const __bf16* ar = &As[(wm * 32 + mt * 16 + l15) * AS_STRIDE];
            bf8 lo = *(const bf8*)(ar + khalf);
            bf8 hi = *(const bf8*)(ar + 16 + khalf);
            #pragma unroll
            for (int e = 0; e < 8; ++e) { afrag[mt][e] = lo[e]; afrag[mt][e + 8] = hi[e]; }
        }
        #pragma unroll
        for (int nt = 0; nt < 2; ++nt) {
            const __bf16* bp = Bt + (size_t)(nblk + wn * 32 + nt * 16 + l15) * kN + k0;
            bf8 lo = *(const bf8*)(bp + khalf);
            bf8 hi = *(const bf8*)(bp + 16 + khalf);
            #pragma unroll
            for (int e = 0; e < 8; ++e) { bfrag[nt][e] = lo[e]; bfrag[nt][e + 8] = hi[e]; }
        }
        #pragma unroll
        for (int mt = 0; mt < 2; ++mt)
            #pragma unroll
            for (int nt = 0; nt < 2; ++nt)
                acc[mt][nt] = __builtin_amdgcn_wmma_f32_16x16x32_bf16(
                    false, afrag[mt], false, bfrag[nt],
                    (short)0, acc[mt][nt], false, false);
        __syncthreads();
    }

    // epilogue per ISA C/D layout: VGPR r -> M=r (lanes 0-15) / r+8 (lanes 16-31)
    const int mh = (lane >> 4) * 8;
    #pragma unroll
    for (int mt = 0; mt < 2; ++mt) {
        #pragma unroll
        for (int nt = 0; nt < 2; ++nt) {
            const int n = nblk + wn * 32 + nt * 16 + l15;
            #pragma unroll
            for (int r = 0; r < 8; ++r) {
                const int m = mblk + wm * 32 + mt * 16 + r + mh;
                float v = acc[mt][nt][r];
                v = (mode == 0) ? v * eival[m] : elu_f(v);
                Cout[(size_t)m * ncols + n] = v;
            }
        }
    }
}

// ================= K4: masked spectral regularization reduction =================
__global__ __launch_bounds__(256)
void reg_kernel(const float* __restrict__ Y,
                const float* __restrict__ eival,
                const int* __restrict__ thres,
                float* __restrict__ acc) {
    __shared__ float sred[256];
    const int i = blockIdx.x * 256 + threadIdx.x;
    float s = 0.f, cnt = 0.f;
    if (i < kN - 1) {
        if ((eival[i] - eival[i + 1]) > 0.01f) {
            cnt = 1.f;
            const float th = (float)thres[0];
            const float* y0 = Y + (size_t)i * 128;
            const float* y1 = Y + (size_t)(i + 1) * 128;
            for (int d = 0; d < kD; ++d) {
                float delta  = fabsf(y0[d]      - y1[d]);
                float delta0 = fabsf(y0[64 + d] - y1[64 + d]);
                float r = delta - th * delta0;
                s += fmaxf(r, 0.f);
            }
        }
    }
    sred[threadIdx.x] = s;
    __syncthreads();
    for (int off = 128; off > 0; off >>= 1) {
        if (threadIdx.x < off) sred[threadIdx.x] += sred[threadIdx.x + off];
        __syncthreads();
    }
    if (threadIdx.x == 0) atomicAdd(&acc[0], sred[0]);
    __syncthreads();
    sred[threadIdx.x] = cnt;
    __syncthreads();
    for (int off = 128; off > 0; off >>= 1) {
        if (threadIdx.x < off) sred[threadIdx.x] += sred[threadIdx.x + off];
        __syncthreads();
    }
    if (threadIdx.x == 0) atomicAdd(&acc[1], sred[0]);
}

// ================= K5: zero scratch accumulators =================
__global__ void zero_kernel(float* __restrict__ sums, float* __restrict__ cnts,
                            float* __restrict__ acc) {
    const int t = blockIdx.x * 256 + threadIdx.x;
    if (t < kC * kFT) sums[t] = 0.f;
    if (t < 16) cnts[t] = 0.f;
    if (t < 8)  acc[t]  = 0.f;
}

// ================= K6: class prototype scatter-sum =================
__global__ __launch_bounds__(256)
void proto_kernel(const float* __restrict__ E, const int* __restrict__ idx,
                  const int* __restrict__ labels,
                  float* __restrict__ sums, float* __restrict__ cnts) {
    const int i = blockIdx.x;
    const int t = threadIdx.x;
    const int lab = labels[i];
    const int row = idx[i];
    atomicAdd(&sums[lab * kFT + t], E[(size_t)row * kFT + t]);
    if (t == 0) atomicAdd(&cnts[lab], 1.f);
}

// ================= K7: normalized prototypes =================
__global__ __launch_bounds__(256)
void pn_kernel(const float* __restrict__ sums, const float* __restrict__ cnts,
               float* __restrict__ pn) {
    __shared__ float red[256];
    __shared__ float nrm;
    const int t = threadIdx.x;
    for (int c = 0; c < kC; ++c) {
        float cn = fmaxf(cnts[c], 1.f);
        float v = sums[c * kFT + t] / cn;
        red[t] = v * v;
        __syncthreads();
        for (int off = 128; off > 0; off >>= 1) {
            if (t < off) red[t] += red[t + off];
            __syncthreads();
        }
        if (t == 0) nrm = fmaxf(sqrtf(red[0]), 1e-8f);
        __syncthreads();
        pn[c * kFT + t] = v / nrm;
        __syncthreads();
    }
}

// ================= K8: cosine similarity + softmax + entropy + reg write =================
__global__ __launch_bounds__(256)
void cos_kernel(const float* __restrict__ E, const int* __restrict__ idx,
                const float* __restrict__ pn, const float* __restrict__ acc,
                float* __restrict__ out) {
    __shared__ float red[256];
    __shared__ float logit[kC];
    __shared__ float qn;
    const int i = blockIdx.x;
    const int t = threadIdx.x;
    const int row = idx[i];
    const float q = E[(size_t)row * kFT + t];
    red[t] = q * q;
    __syncthreads();
    for (int off = 128; off > 0; off >>= 1) {
        if (t < off) red[t] += red[t + off];
        __syncthreads();
    }
    if (t == 0) qn = fmaxf(sqrtf(red[0]), 1e-8f);
    __syncthreads();
    const float qv = q / qn;
    for (int c = 0; c < kC; ++c) {
        red[t] = qv * pn[c * kFT + t];
        __syncthreads();
        for (int off = 128; off > 0; off >>= 1) {
            if (t < off) red[t] += red[t + off];
            __syncthreads();
        }
        if (t == 0) logit[c] = red[0];
        __syncthreads();
    }
    if (t == 0) {
        float mx = logit[0];
        for (int c = 1; c < kC; ++c) mx = fmaxf(mx, logit[c]);
        float e[kC], sm = 0.f;
        for (int c = 0; c < kC; ++c) { e[c] = __expf(logit[c] - mx); sm += e[c]; }
        float ent = 0.f;
        for (int c = 0; c < kC; ++c) {
            float p = e[c] / sm;
            out[(size_t)i * kC + c] = p;
            ent -= p * __logf(p + 1e-8f);
        }
        out[(size_t)kNIDX * kC + i] = ent;
        if (i == 0) out[(size_t)kNIDX * kC + kNIDX] = acc[0] / (fmaxf(acc[1], 1e-20f) * (float)kD);
    }
}

// ================= launch =================
extern "C" void kernel_launch(void* const* d_in, const int* in_sizes, int n_in,
                              void* d_out, int out_size, void* d_ws, size_t ws_size,
                              hipStream_t stream) {
    (void)in_sizes; (void)n_in; (void)out_size; (void)ws_size;
    const float* eivec     = (const float*)d_in[0];
    const float* eival     = (const float*)d_in[1];
    const float* features  = (const float*)d_in[2];
    const float* adj       = (const float*)d_in[3];
    const int*   idx       = (const int*)  d_in[4];
    const int*   labels    = (const int*)  d_in[5];
    const float* texttoken = (const float*)d_in[6];
    const float* wp        = (const float*)d_in[7];
    const float* r_w1      = (const float*)d_in[8];
    const float* r_b1      = (const float*)d_in[9];
    const float* r_w2      = (const float*)d_in[10];
    const float* r_b2      = (const float*)d_in[11];
    const float* down      = (const float*)d_in[12];
    const float* comb      = (const float*)d_in[13];
    const float* w_gcn     = (const float*)d_in[14];
    const int*   thres     = (const int*)  d_in[15];

    char* ws = (char*)d_ws;
    float*  coef = (float*) (ws + OFF_COEF);
    float*  F1   = (float*) (ws + OFF_F1);
    __bf16* Bt   = (__bf16*)(ws + OFF_BT);
    __bf16* Ht   = (__bf16*)(ws + OFF_HT);
    float*  Y    = (float*) (ws + OFF_Y);
    float*  E    = (float*) (ws + OFF_E);
    float*  sums = (float*) (ws + OFF_SUMS);
    float*  cnts = (float*) (ws + OFF_CNT);
    float*  pn   = (float*) (ws + OFF_PN);
    float*  acc  = (float*) (ws + OFF_ACC);
    float*  out  = (float*)d_out;

    zero_kernel<<<dim3((kC * kFT + 255) / 256), dim3(256), 0, stream>>>(sums, cnts, acc);
    prompt_kernel<<<dim3(1), dim3(64), 0, stream>>>(texttoken, wp, r_w1, r_b1, r_w2, r_b2,
                                                    down, comb, coef);
    features1_kernel<<<dim3(kN / 64), dim3(256), 0, stream>>>(features, coef, F1, Bt);
    hgcn_kernel<<<dim3(kN / 64, kFT / 64), dim3(256), 0, stream>>>(F1, w_gcn, Ht);
    // GEMM1: Y = diag(eival) * (eivec @ [features1 | features]),  N-cols = 128
    gemm_bf16_kernel<<<dim3(kN / 128, 128 / 64), dim3(256), 0, stream>>>(
        eivec, Bt, Y, 128, eival, 0);
    // GEMM2: E = elu(adj @ Hgcn),  N-cols = 256
    gemm_bf16_kernel<<<dim3(kN / 128, kFT / 64), dim3(256), 0, stream>>>(
        adj, Ht, E, kFT, eival, 1);
    reg_kernel<<<dim3((kN - 1 + 255) / 256), dim3(256), 0, stream>>>(Y, eival, thres, acc);
    proto_kernel<<<dim3(kNIDX), dim3(256), 0, stream>>>(E, idx, labels, sums, cnts);
    pn_kernel<<<dim3(1), dim3(256), 0, stream>>>(sums, cnts, pn);
    cos_kernel<<<dim3(kNIDX), dim3(256), 0, stream>>>(E, idx, pn, acc, out);
}
